// SparseMultiHeadAttention_61632780697693
// MI455X (gfx1250) — compile-verified, matrix operand order checked by reading, not run on hardware
//
#include <hip/hip_runtime.h>

// ---------------------------------------------------------------------------
// Sparse multi-head attention for MI455X (gfx1250).
// Matrix math: v_wmma_f32_16x16x32_bf16 (f32 accum).
// LDS staging in the GEMMs: global_load_async_to_lds_b128 (ASYNCcnt),
// double-buffered so DMA of tile i+1 overlaps WMMA of tile i.
// ---------------------------------------------------------------------------

typedef __attribute__((ext_vector_type(16))) __bf16 bf16x16;
typedef __attribute__((ext_vector_type(8)))  __bf16 bf16x8;
typedef __attribute__((ext_vector_type(8)))  float  f32x8;

#define S_LEN  4096
#define D_DIM  1024
#define N_HEAD 16
#define DKH    64
#define WIN    256
#define NGLOB  16
#define LDSPITCH 40   // 32 + 8 pad: 16B-aligned rows, spread banks

typedef __attribute__((address_space(3))) void lds_void_t;

__device__ __forceinline__ bf16x8 ld8(const __bf16* p) {
  return *(const bf16x8*)p;
}
__device__ __forceinline__ bf16x16 cmb(bf16x8 lo, bf16x8 hi) {
  return __builtin_shufflevector(lo, hi, 0, 1, 2, 3, 4, 5, 6, 7,
                                          8, 9, 10, 11, 12, 13, 14, 15);
}
__device__ __forceinline__ f32x8 wmma_bf16(bf16x16 a, bf16x16 b, f32x8 c) {
  return __builtin_amdgcn_wmma_f32_16x16x32_bf16(
      false, a, false, b, (short)0, c, false, false);
}

// Async global -> LDS copy, 16 bytes per lane. Tracked with ASYNCcnt.
__device__ __forceinline__ void async_b128(void* lds_dst, const void* gsrc) {
  unsigned loff = (unsigned)(unsigned long long)(lds_void_t*)lds_dst;
  asm volatile("global_load_async_to_lds_b128 %0, %1, off"
               :: "v"(loff), "v"(gsrc)
               : "memory");
}
__device__ __forceinline__ void wait_async0() {
  asm volatile("s_wait_asynccnt 0x0" ::: "memory");
}

// ---------------------------------------------------------------------------
// fp32 -> bf16 elementwise (activations)
// ---------------------------------------------------------------------------
__global__ void cvt_f32_bf16(const float* __restrict__ in,
                             __bf16* __restrict__ out, int n) {
  int i = blockIdx.x * blockDim.x + threadIdx.x;
  int stride = gridDim.x * blockDim.x;
  for (; i < n; i += stride) out[i] = (__bf16)in[i];
}

// ---------------------------------------------------------------------------
// fp32 [K][N] -> bf16 transposed [N][K]   (weights, K = N = 1024)
// ---------------------------------------------------------------------------
__global__ __launch_bounds__(256)
void cvt_t_f32_bf16(const float* __restrict__ in, __bf16* __restrict__ out) {
  __shared__ float tile[32][33];
  const int bx = blockIdx.x * 32, by = blockIdx.y * 32;
  const int tx = threadIdx.x, ty = threadIdx.y;  // (32, 8)
#pragma unroll
  for (int r = 0; r < 32; r += 8)
    tile[ty + r][tx] = in[(size_t)(by + ty + r) * D_DIM + bx + tx];
  __syncthreads();
#pragma unroll
  for (int r = 0; r < 32; r += 8)
    out[(size_t)(bx + ty + r) * D_DIM + by + tx] = (__bf16)tile[tx][ty + r];
}

// ---------------------------------------------------------------------------
// bf16 [rows][cols] -> bf16 [cols][rows]   (V -> V^T)
// ---------------------------------------------------------------------------
__global__ __launch_bounds__(256)
void transpose_bf16(const __bf16* __restrict__ in, __bf16* __restrict__ out,
                    int rows, int cols) {
  __shared__ __bf16 tile[32][33];
  const int bx = blockIdx.x * 32, by = blockIdx.y * 32;
  const int tx = threadIdx.x, ty = threadIdx.y;  // (32, 8)
#pragma unroll
  for (int r = 0; r < 32; r += 8)
    tile[ty + r][tx] = in[(size_t)(by + ty + r) * cols + bx + tx];
  __syncthreads();
#pragma unroll
  for (int r = 0; r < 32; r += 8)
    out[(size_t)(bx + ty + r) * rows + by + tx] = tile[tx][ty + r];
}

// ---------------------------------------------------------------------------
// Tiled bf16 WMMA GEMM: C[4096][N] = A[4096x1024] * B[1024xN] + bias,
// with B supplied TRANSPOSED (BT[N][1024]) so both LDS tiles are straight
// row copies staged with async global->LDS b128, double buffered.
// Block tile 128x128, 8 waves, wave tile 32x64.
// Fragment layout (16-bit): lane L -> row/col = L%16, half = L/16,
//   elems 0..7 -> k = half*8+e ; elems 8..15 -> k = 16+half*8+e.
// ---------------------------------------------------------------------------
template <bool OUT_BF16>
__global__ __launch_bounds__(256)
void gemm_bf16(const __bf16* __restrict__ A, const __bf16* __restrict__ BT,
               const float* __restrict__ bias, void* __restrict__ outp,
               int N) {
  __shared__ alignas(16) __bf16 lA[2][128 * LDSPITCH];
  __shared__ alignas(16) __bf16 lB[2][128 * LDSPITCH];

  const int tid = threadIdx.x;
  const int lane = tid & 31;
  const int w  = tid >> 5;      // 0..7
  const int wm = w >> 1;        // 0..3
  const int wn = w & 1;         // 0..1
  const int lr = lane & 15;
  const int hl = lane >> 4;
  const int bm = blockIdx.y * 128;
  const int bn = blockIdx.x * 128;
  const int trow = tid >> 1, tcol = (tid & 1) * 16;  // 128 rows x 32 cols

  const __bf16* gA = A  + (size_t)(bm + trow) * D_DIM + tcol;
  const __bf16* gB = BT + (size_t)(bn + trow) * D_DIM + tcol;
  __bf16* sA0 = &lA[0][trow * LDSPITCH + tcol];
  __bf16* sA1 = &lA[1][trow * LDSPITCH + tcol];
  __bf16* sB0 = &lB[0][trow * LDSPITCH + tcol];
  __bf16* sB1 = &lB[1][trow * LDSPITCH + tcol];

  f32x8 acc[2][4];
#pragma unroll
  for (int mt = 0; mt < 2; ++mt)
#pragma unroll
    for (int nt = 0; nt < 4; ++nt) acc[mt][nt] = (f32x8)0.0f;

  // prologue: stage tile 0 into buffer 0
  async_b128(sA0, gA);
  async_b128(sA0 + 8, gA + 8);
  async_b128(sB0, gB);
  async_b128(sB0 + 8, gB + 8);

  for (int i = 0; i < D_DIM / 32; ++i) {
    wait_async0();     // my async copies into buf (i&1) are in LDS
    __syncthreads();   // everyone's copies done; prior reads of other buf done

    if (i + 1 < D_DIM / 32) {  // stage next tile into the other buffer
      const int k1 = (i + 1) * 32;
      __bf16* dA = (i & 1) ? sA0 : sA1;
      __bf16* dB = (i & 1) ? sB0 : sB1;
      async_b128(dA, gA + k1);
      async_b128(dA + 8, gA + k1 + 8);
      async_b128(dB, gB + k1);
      async_b128(dB + 8, gB + k1 + 8);
    }

    const __bf16* bufA = lA[i & 1];
    const __bf16* bufB = lB[i & 1];
    bf16x16 af[2], bfr[4];
#pragma unroll
    for (int mt = 0; mt < 2; ++mt) {
      const __bf16* p = &bufA[(wm * 32 + mt * 16 + lr) * LDSPITCH];
      af[mt] = cmb(ld8(p + hl * 8), ld8(p + 16 + hl * 8));
    }
#pragma unroll
    for (int nt = 0; nt < 4; ++nt) {
      const __bf16* p = &bufB[(wn * 64 + nt * 16 + lr) * LDSPITCH];
      bfr[nt] = cmb(ld8(p + hl * 8), ld8(p + 16 + hl * 8));
    }
#pragma unroll
    for (int mt = 0; mt < 2; ++mt)
#pragma unroll
      for (int nt = 0; nt < 4; ++nt)
        acc[mt][nt] = wmma_bf16(af[mt], bfr[nt], acc[mt][nt]);
  }

  // Epilogue: C layout row m = v + 8*half, col = lane%16
#pragma unroll
  for (int mt = 0; mt < 2; ++mt) {
#pragma unroll
    for (int nt = 0; nt < 4; ++nt) {
      const int col = bn + wn * 64 + nt * 16 + lr;
      const float bv = bias[col];
#pragma unroll
      for (int v = 0; v < 8; ++v) {
        const int row = bm + wm * 32 + mt * 16 + v + 8 * hl;
        const float val = acc[mt][nt][v] + bv;
        if constexpr (OUT_BF16)
          ((__bf16*)outp)[(size_t)row * N + col] = (__bf16)val;
        else
          ((float*)outp)[(size_t)row * N + col] = val;
      }
    }
  }
}

// ---------------------------------------------------------------------------
// Flash-style sparse attention. One wave per (16-query block, head).
// Streams 32-key pair-tiles: global tile (kb=0) first, then causal window.
// V is pre-transposed (VpT[D][S]) so P*V B-fragments are contiguous loads.
// ---------------------------------------------------------------------------
__global__ __launch_bounds__(32)
void attn_sparse(const __bf16* __restrict__ Qp, const __bf16* __restrict__ Kp,
                 const __bf16* __restrict__ VpT, __bf16* __restrict__ Op) {
  __shared__ alignas(16) __bf16 lP[16 * LDSPITCH];  // probs (C->A relayout)

  const int lane = threadIdx.x & 31;
  const int lr = lane & 15;
  const int hl = lane >> 4;
  const int qstart = blockIdx.x * 16;
  const int head = blockIdx.y;

  const __bf16* qb = Qp + (size_t)(qstart + lr) * D_DIM + head * DKH;
  const bf16x16 aq0 = cmb(ld8(qb + hl * 8),      ld8(qb + 16 + hl * 8));
  const bf16x16 aq1 = cmb(ld8(qb + 32 + hl * 8), ld8(qb + 48 + hl * 8));

  f32x8 acc[4];
#pragma unroll
  for (int nt = 0; nt < 4; ++nt) acc[nt] = (f32x8)0.0f;
  float mrow[8], lrow[8];
#pragma unroll
  for (int v = 0; v < 8; ++v) { mrow[v] = -1e30f; lrow[v] = 0.0f; }

  int lo = qstart - WIN;
  if (lo < 0) lo = 0;
  lo &= ~31;
  const int nwin = ((qstart + 15) - lo) / 32 + 1;
  const int ntiles = nwin + (lo > 0 ? 1 : 0);

  for (int t = 0; t < ntiles; ++t) {
    const int kb = (lo > 0) ? ((t == 0) ? 0 : lo + (t - 1) * 32)
                            : lo + t * 32;
    if (t + 1 < ntiles) {  // warm L2 for next tile's K rows
      const int nkb = (lo > 0) ? ((t == 0) ? lo : kb + 32) : kb + 32;
      __builtin_prefetch(
          (const void*)(Kp + (size_t)(nkb + lane) * D_DIM + head * DKH), 0, 1);
    }

    // ---- scores: two 16x16 tiles over DK=64 (two K=32 WMMAs each) ----
    f32x8 sc[2];
#pragma unroll
    for (int sub = 0; sub < 2; ++sub) {
      const __bf16* kbase =
          Kp + (size_t)(kb + sub * 16 + lr) * D_DIM + head * DKH;
      const bf16x16 bk0 = cmb(ld8(kbase + hl * 8),      ld8(kbase + 16 + hl * 8));
      const bf16x16 bk1 = cmb(ld8(kbase + 32 + hl * 8), ld8(kbase + 48 + hl * 8));
      f32x8 z = (f32x8)0.0f;
      z = wmma_bf16(aq0, bk0, z);
      z = wmma_bf16(aq1, bk1, z);
      sc[sub] = z;
    }

    // ---- mask + online softmax ----
    float pa[8], pb[8], cr[8];
#pragma unroll
    for (int v = 0; v < 8; ++v) {
      const int i = qstart + v + 8 * hl;
      const int j0 = kb + lr;
      const int j1 = kb + 16 + lr;
      float s0 = sc[0][v] * 0.125f;  // 1/sqrt(DK)
      float s1 = sc[1][v] * 0.125f;
      const bool ok0 = (j0 <= i) && ((i - j0) <= WIN || j0 < NGLOB);
      const bool ok1 = (j1 <= i) && ((i - j1) <= WIN || j1 < NGLOB);
      s0 = ok0 ? s0 : -1e30f;
      s1 = ok1 ? s1 : -1e30f;
      float rm = fmaxf(s0, s1);
      rm = fmaxf(rm, __shfl_xor(rm, 1, 32));
      rm = fmaxf(rm, __shfl_xor(rm, 2, 32));
      rm = fmaxf(rm, __shfl_xor(rm, 4, 32));
      rm = fmaxf(rm, __shfl_xor(rm, 8, 32));
      const float mn = fmaxf(mrow[v], rm);
      const float corr = __expf(mrow[v] - mn);
      mrow[v] = mn;
      const float p0 = __expf(s0 - mn);
      const float p1 = __expf(s1 - mn);
      float rs = p0 + p1;
      rs += __shfl_xor(rs, 1, 32);
      rs += __shfl_xor(rs, 2, 32);
      rs += __shfl_xor(rs, 4, 32);
      rs += __shfl_xor(rs, 8, 32);
      lrow[v] = lrow[v] * corr + rs;
      pa[v] = p0; pb[v] = p1; cr[v] = corr;
    }
#pragma unroll
    for (int nt = 0; nt < 4; ++nt)
#pragma unroll
      for (int v = 0; v < 8; ++v) acc[nt][v] *= cr[v];

    // ---- relayout P (C layout -> A layout) through LDS ----
    __syncthreads();
#pragma unroll
    for (int v = 0; v < 8; ++v) {
      const int m = v + 8 * hl;
      lP[m * LDSPITCH + lr]      = (__bf16)pa[v];
      lP[m * LDSPITCH + 16 + lr] = (__bf16)pb[v];
    }
    __syncthreads();
    const __bf16* pp = &lP[lr * LDSPITCH];
    const bf16x16 pf = cmb(ld8(pp + hl * 8), ld8(pp + 16 + hl * 8));

    // ---- P (16x32) x V (32x64): B-frags straight from V^T ----
#pragma unroll
    for (int nt = 0; nt < 4; ++nt) {
      const __bf16* vp =
          VpT + (size_t)(head * DKH + nt * 16 + lr) * S_LEN + kb;
      const bf16x16 bv = cmb(ld8(vp + hl * 8), ld8(vp + 16 + hl * 8));
      acc[nt] = wmma_bf16(pf, bv, acc[nt]);
    }
  }

  // ---- normalize + store bf16 [S][D] ----
  float invl[8];
#pragma unroll
  for (int v = 0; v < 8; ++v) invl[v] = 1.0f / lrow[v];
#pragma unroll
  for (int nt = 0; nt < 4; ++nt) {
#pragma unroll
    for (int v = 0; v < 8; ++v) {
      const int row = qstart + v + 8 * hl;
      const int col = head * DKH + nt * 16 + lr;
      Op[(size_t)row * D_DIM + col] = (__bf16)(acc[nt][v] * invl[v]);
    }
  }
}

// ---------------------------------------------------------------------------
// Host-side launcher
// ---------------------------------------------------------------------------
extern "C" void kernel_launch(void* const* d_in, const int* in_sizes, int n_in,
                              void* d_out, int out_size, void* d_ws,
                              size_t ws_size, hipStream_t stream) {
  (void)in_sizes; (void)n_in; (void)out_size; (void)ws_size;
  const float* query = (const float*)d_in[0];
  const float* key   = (const float*)d_in[1];
  const float* value = (const float*)d_in[2];
  const float* Wq = (const float*)d_in[3];
  const float* bq = (const float*)d_in[4];
  const float* Wk = (const float*)d_in[5];
  const float* bk = (const float*)d_in[6];
  const float* Wv = (const float*)d_in[7];
  const float* bv = (const float*)d_in[8];
  const float* Wo = (const float*)d_in[9];
  const float* bo = (const float*)d_in[10];

  constexpr size_t SB = (size_t)S_LEN * D_DIM;
  constexpr size_t WB = (size_t)D_DIM * D_DIM;

  __bf16* w = (__bf16*)d_ws;
  __bf16* Xq  = w;             // [S][D] bf16 activations
  __bf16* Xk  = w + SB;
  __bf16* Xv  = w + 2 * SB;
  __bf16* WqT = w + 3 * SB;    // [N][K] bf16 transposed weights
  __bf16* WkT = WqT + WB;
  __bf16* WvT = WqT + 2 * WB;
  __bf16* WoT = WqT + 3 * WB;
  __bf16* Qp  = WqT + 4 * WB;  // [S][H*DK] projections
  __bf16* Kp  = Qp + SB;
  __bf16* Vp  = Qp + 2 * SB;
  __bf16* VpT = Xk;            // alias: Xk dead after K projection
  __bf16* Ap  = Xq;            // alias: Xq dead after Q projection
  // total ws: 6*SB + 4*WB bf16 = ~56 MiB

  cvt_f32_bf16<<<2048, 256, 0, stream>>>(query, Xq, (int)SB);
  cvt_f32_bf16<<<2048, 256, 0, stream>>>(key,   Xk, (int)SB);
  cvt_f32_bf16<<<2048, 256, 0, stream>>>(value, Xv, (int)SB);

  const dim3 tb(32, 8);
  cvt_t_f32_bf16<<<dim3(32, 32), tb, 0, stream>>>(Wq, WqT);
  cvt_t_f32_bf16<<<dim3(32, 32), tb, 0, stream>>>(Wk, WkT);
  cvt_t_f32_bf16<<<dim3(32, 32), tb, 0, stream>>>(Wv, WvT);
  cvt_t_f32_bf16<<<dim3(32, 32), tb, 0, stream>>>(Wo, WoT);

  const dim3 gb(D_DIM / 128, S_LEN / 128);  // (8, 32)
  gemm_bf16<true><<<gb, 256, 0, stream>>>(Xq, WqT, bq, Qp, D_DIM);
  gemm_bf16<true><<<gb, 256, 0, stream>>>(Xk, WkT, bk, Kp, D_DIM);
  gemm_bf16<true><<<gb, 256, 0, stream>>>(Xv, WvT, bv, Vp, D_DIM);

  transpose_bf16<<<dim3(D_DIM / 32, S_LEN / 32), tb, 0, stream>>>(
      Vp, VpT, S_LEN, D_DIM);

  attn_sparse<<<dim3(S_LEN / 16, N_HEAD), 32, 0, stream>>>(Qp, Kp, VpT, Ap);

  gemm_bf16<false><<<gb, 256, 0, stream>>>(Ap, WoT, bo, d_out, D_DIM);
}
